// _STEEGAttention_76381698392433
// MI455X (gfx1250) — matrix operand order repaired and checked by reading, not compiled, not run on hardware
//
#include <hip/hip_runtime.h>
#include <cstdint>
#include <type_traits>

// ---------------------------------------------------------------------------
// CDNA5 bf16 WMMA attention block, TDM-staged:
//   0) one-shot bf16 conversion of x, and transpose+convert of W_qkv/W_proj
//   1) QKV GEMM  (bf16 A x bf16 W^T -> bf16),   TDM double-buffered tiles
//   2) flash attention per (b,h), bf16 WMMA,    TDM-staged K tiles
//   3) proj GEMM (bf16 A x bf16 W^T -> fp32),   TDM double-buffered tiles
// ---------------------------------------------------------------------------

typedef __attribute__((ext_vector_type(16))) __bf16   v16bf;
typedef __attribute__((ext_vector_type(8)))  float    v8f;
typedef __attribute__((ext_vector_type(8)))  uint32_t v8u;
typedef __attribute__((ext_vector_type(4)))  uint32_t su4;
typedef __attribute__((ext_vector_type(8)))  uint32_t su8;

static __device__ __forceinline__ uint16_t f32_to_bf16(float f) {
  uint32_t u = __builtin_bit_cast(uint32_t, f);
  u += 0x7FFFu + ((u >> 16) & 1u);           // round-to-nearest-even
  return (uint16_t)(u >> 16);
}
static __device__ __forceinline__ uint32_t pack_bf16x2(float a, float b) {
  return (uint32_t)f32_to_bf16(a) | ((uint32_t)f32_to_bf16(b) << 16);
}

static __device__ __forceinline__ v16bf frag_from(uint4 lo, uint4 hi4) {
  v8u u;
  u[0] = lo.x;  u[1] = lo.y;  u[2] = lo.z;  u[3] = lo.w;
  u[4] = hi4.x; u[5] = hi4.y; u[6] = hi4.z; u[7] = hi4.w;
  return __builtin_bit_cast(v16bf, u);
}

// A fragment (16x32 bf16): row = lane&15; halves {kb..kb+7, kb+16..kb+23},
// kb = (lane>>4)*8.  `row` points at 32 contiguous bf16 K values of this row.
static __device__ __forceinline__ v16bf load_fragA(const uint16_t* row, int hi) {
  const int kb = hi * 8;
  uint4 lo  = *(const uint4*)(row + kb);
  uint4 hi4 = *(const uint4*)(row + kb + 16);
  return frag_from(lo, hi4);
}
// B fragment (32x16 bf16): col = lane&15; halves {kb..kb+7, kb+16..kb+23},
// kb = (lane>>4)*8. `colk` -> 32 contiguous K halves of this column.
static __device__ __forceinline__ v16bf load_fragB(const uint16_t* colk, int hi) {
  const int kb = hi * 8;
  uint4 lo  = *(const uint4*)(colk + kb);
  uint4 hi4 = *(const uint4*)(colk + kb + 16);
  return frag_from(lo, hi4);
}

static __device__ __forceinline__ v8f wmma_bf16(v16bf a, v16bf b, v8f c) {
  return __builtin_amdgcn_wmma_f32_16x16x32_bf16(
      false, a, false, b, (short)0, c, false, false);
}

static __device__ __forceinline__ uint32_t lds_off(const void* p) {
  // flat LDS address: addr[31:0] = byte offset into LDS space
  return (uint32_t)(uintptr_t)p;
}

// ---------------------------------------------------------------------------
// Tensor Data Mover: 2D bf16 tile load, global -> LDS.
// D# group0: count=1 | lds_addr | global_addr | type=2
// D# group1: data_size=2B, tensor_dim0/1 = tile dims (no OOB), dim0 stride.
// Groups 2/3 zero (tile_dim2..4 = 0 => unused).  Issued once per wave;
// completion tracked with TENSORcnt (in-order per wave).
// ---------------------------------------------------------------------------
static __device__ __forceinline__ void tdm_load_2d(
    uint32_t lds_byte_off, const void* gptr,
    uint32_t tile0 /*elems per row*/, uint32_t tile1 /*rows*/,
    uint32_t stride_elems /*row stride in elements*/) {
  const uint64_t ga = (uint64_t)(uintptr_t)gptr;
  su4 g0;
  g0[0] = (uint32_t)__builtin_amdgcn_readfirstlane(1u);  // count=1
  g0[1] = (uint32_t)__builtin_amdgcn_readfirstlane(lds_byte_off);
  g0[2] = (uint32_t)__builtin_amdgcn_readfirstlane((uint32_t)ga);
  g0[3] = (uint32_t)__builtin_amdgcn_readfirstlane((uint32_t)(ga >> 32) | (2u << 30));
  su8 g1;
  g1[0] = (uint32_t)__builtin_amdgcn_readfirstlane(1u << 16);          // data_size=2B
  g1[1] = (uint32_t)__builtin_amdgcn_readfirstlane(tile0 << 16);       // tensor_dim0 lo
  g1[2] = (uint32_t)__builtin_amdgcn_readfirstlane((tile0 >> 16) | (tile1 << 16));
  g1[3] = (uint32_t)__builtin_amdgcn_readfirstlane((tile1 >> 16) | (tile0 << 16)); // tile_dim0
  g1[4] = (uint32_t)__builtin_amdgcn_readfirstlane(tile1 & 0xFFFFu);   // tile_dim1
  g1[5] = (uint32_t)__builtin_amdgcn_readfirstlane(stride_elems);      // dim0_stride lo
  g1[6] = (uint32_t)__builtin_amdgcn_readfirstlane(0u);
  g1[7] = (uint32_t)__builtin_amdgcn_readfirstlane(0u);
  su4 gz;
  gz[0] = gz[1] = gz[2] = gz[3] = (uint32_t)__builtin_amdgcn_readfirstlane(0u);
  asm volatile("tensor_load_to_lds %0, %1, %2, %3"
               :: "s"(g0), "s"(g1), "s"(gz), "s"(gz)
               : "memory");
}

// ---------------------------------------------------------------------------
// One-shot converters (bandwidth-trivial prepass, amortized over tile reuse)
// ---------------------------------------------------------------------------
__global__ __launch_bounds__(256) void convert_f32_bf16(
    const float* __restrict__ in, uint16_t* __restrict__ out, int n4) {
  const int i = blockIdx.x * 256 + threadIdx.x;
  if (i < n4) {
    const float4 f = ((const float4*)in)[i];
    uint2 p;
    p.x = pack_bf16x2(f.x, f.y);
    p.y = pack_bf16x2(f.z, f.w);
    ((uint2*)out)[i] = p;
  }
}

// Wt[n][k] = bf16(W[k][n]);  tiled transpose through LDS
__global__ __launch_bounds__(256) void transpose_f32_bf16(
    const float* __restrict__ W, uint16_t* __restrict__ Wt, int K, int N) {
  __shared__ float tile[32][33];
  const int tx = threadIdx.x & 31, ty = threadIdx.x >> 5;   // 32x8
  const int n0 = blockIdx.x * 32, k0 = blockIdx.y * 32;
#pragma unroll
  for (int j = 0; j < 32; j += 8)
    tile[ty + j][tx] = W[(size_t)(k0 + ty + j) * N + n0 + tx];
  __syncthreads();
#pragma unroll
  for (int j = 0; j < 32; j += 8)
    Wt[(size_t)(n0 + ty + j) * K + k0 + tx] = f32_to_bf16(tile[tx][ty + j]);
}

// ---------------------------------------------------------------------------
// GEMM with fused bias: C[M,N] = A[M,K] * Wt[N,K]^T + bias[N]   (all-bf16 in)
// Block = 256 threads (8 waves), tile 128x64, K stepped by 64.
// Both input tiles staged by TDM, double-buffered: wave 0 issues next tile's
// descriptors, s_wait_tensorcnt(2) guarantees current tile resident.
// ---------------------------------------------------------------------------
template <typename OT>
__global__ __launch_bounds__(256) void gemm_bias_wmma_tdm(
    const uint16_t* __restrict__ A,    // [M][K] bf16 row-major
    const uint16_t* __restrict__ Wt,   // [N][K] bf16 (pre-transposed)
    const float* __restrict__ bias, OT* __restrict__ C,
    int M, int N, int K) {
  __shared__ __align__(16) uint16_t lA[2][128][64];   // 32 KB
  __shared__ __align__(16) uint16_t lW[2][64][64];    // 16 KB

  const int t = threadIdx.x;
  const int wave = t >> 5, lane = t & 31;
  const int n = lane & 15, hi = lane >> 4;
  const int mt = blockIdx.x * 128, nt = blockIdx.y * 64;

  v8f acc[4] = {};

  if (wave == 0) {
    tdm_load_2d(lds_off(&lA[0][0][0]), A + (size_t)mt * K, 64, 128, (uint32_t)K);
    tdm_load_2d(lds_off(&lW[0][0][0]), Wt + (size_t)nt * K, 64, 64, (uint32_t)K);
  }

  for (int k0 = 0; k0 < K; k0 += 64) {
    const int cur = (k0 >> 6) & 1;
    __syncthreads();                 // all waves done reading buf[1-cur]
    if (wave == 0) {
      if (k0 + 64 < K) {
        tdm_load_2d(lds_off(&lA[1 - cur][0][0]),
                    A + (size_t)mt * K + (k0 + 64), 64, 128, (uint32_t)K);
        tdm_load_2d(lds_off(&lW[1 - cur][0][0]),
                    Wt + (size_t)nt * K + (k0 + 64), 64, 64, (uint32_t)K);
        __builtin_amdgcn_s_wait_tensorcnt(2);   // oldest 2 (current tile) done
      } else {
        __builtin_amdgcn_s_wait_tensorcnt(0);
      }
    }
    __syncthreads();                 // publish buf[cur] to all waves

    const v16bf a0 = load_fragA(&lA[cur][wave * 16 + n][0], hi);
    const v16bf a1 = load_fragA(&lA[cur][wave * 16 + n][32], hi);
#pragma unroll
    for (int f = 0; f < 4; ++f) {
      acc[f] = wmma_bf16(a0, load_fragB(&lW[cur][f * 16 + n][0], hi), acc[f]);
      acc[f] = wmma_bf16(a1, load_fragB(&lW[cur][f * 16 + n][32], hi), acc[f]);
    }
  }

  // epilogue: C layout lane(L) VGPR(r) -> row r+(L>>4)*8, col L&15
#pragma unroll
  for (int f = 0; f < 4; ++f) {
    const int col = nt + f * 16 + n;
    const float bv = bias[col];
#pragma unroll
    for (int r = 0; r < 8; ++r) {
      const int row = mt + wave * 16 + r + hi * 8;
      const float v = acc[f][r] + bv;
      if constexpr (std::is_same_v<OT, float>)
        C[(size_t)row * N + col] = v;
      else
        C[(size_t)row * N + col] = f32_to_bf16(v);
    }
  }
}

// ---------------------------------------------------------------------------
// Flash attention, bf16 WMMA. Grid (qchunk=8, head=12, batch=8), 256 threads.
// K tile staged by TDM (wave 0); V tile transposed into LDS by all threads.
// qkv layout: [B*T, 2304] bf16 (Q at col h*64, K at 768+h*64, V at 1536+h*64).
// obuf layout: [B*T, 768] bf16.
// ---------------------------------------------------------------------------
__global__ __launch_bounds__(256) void flash_attn_wmma(
    const uint16_t* __restrict__ qkv, uint16_t* __restrict__ obuf) {
  constexpr int Tq = 1024, TD3 = 2304, HD = 64;
  constexpr float SCALE = 0.125f;  // 64^-0.5

  __shared__ __align__(16) uint16_t lK[32][64];      // keys row-major [k][d]
  __shared__ __align__(16) uint16_t lVt[64][32];     // V transposed   [d][k]
  __shared__ __align__(16) uint16_t lP[8][16][32];   // per-wave P patch

  const int t = threadIdx.x;
  const int wave = t >> 5, lane = t & 31;
  const int n = lane & 15, hi = lane >> 4;
  const int qc = blockIdx.x, h = blockIdx.y, b = blockIdx.z;
  const int qbase = qc * 128 + wave * 16;

  // Q fragments: row = qbase + n, d split 0..31 / 32..63 (held whole kernel)
  const uint16_t* qrow = qkv + (size_t)(b * Tq + qbase + n) * TD3 + h * HD;
  const v16bf qf0 = load_fragA(qrow, hi);
  const v16bf qf1 = load_fragA(qrow + 32, hi);

  v8f oacc[4] = {};
  float mrow[8], lrow[8];
#pragma unroll
  for (int r = 0; r < 8; ++r) { mrow[r] = -1e30f; lrow[r] = 0.f; }

  const int koff = 768 + h * HD, voff = 1536 + h * HD;
  const int skey = t >> 3, sch = t & 7;  // V staging: key row, 8-half chunk

  for (int kt = 0; kt < Tq; kt += 32) {
    __syncthreads();   // previous tile fully consumed
    if (wave == 0)     // K tile [32 x 64] halves, row stride 2304, via TDM
      tdm_load_2d(lds_off(&lK[0][0]),
                  qkv + (size_t)(b * Tq + kt) * TD3 + koff, 64, 32, TD3);
    {  // V tile: load row-major, scatter transposed into lVt[d][k]
      const uint4 vv = *((const uint4*)(qkv + (size_t)(b * Tq + kt + skey) * TD3 + voff) + sch);
      const int d0 = sch * 8;
      lVt[d0 + 0][skey] = (uint16_t)(vv.x);  lVt[d0 + 1][skey] = (uint16_t)(vv.x >> 16);
      lVt[d0 + 2][skey] = (uint16_t)(vv.y);  lVt[d0 + 3][skey] = (uint16_t)(vv.y >> 16);
      lVt[d0 + 4][skey] = (uint16_t)(vv.z);  lVt[d0 + 5][skey] = (uint16_t)(vv.z >> 16);
      lVt[d0 + 6][skey] = (uint16_t)(vv.w);  lVt[d0 + 7][skey] = (uint16_t)(vv.w >> 16);
    }
    if (wave == 0) __builtin_amdgcn_s_wait_tensorcnt(0);
    __syncthreads();

    // --- S = Q K^T for 16q x 32k (two 16-col subtiles, K-dim = 64) ---
    v8f s0 = {}, s1 = {};
    {
      const uint16_t* kc0 = &lK[n][0];        // lane's key column (keys 0..15)
      s0 = wmma_bf16(qf0, load_fragB(kc0, hi), s0);
      s0 = wmma_bf16(qf1, load_fragB(kc0 + 32, hi), s0);
      const uint16_t* kc1 = &lK[16 + n][0];   // keys 16..31
      s1 = wmma_bf16(qf0, load_fragB(kc1, hi), s1);
      s1 = wmma_bf16(qf1, load_fragB(kc1 + 32, hi), s1);
    }

    // --- online softmax: lane holds rows r+(lane>>4)*8; reduce over 16 lanes
#pragma unroll
    for (int r = 0; r < 8; ++r) {
      const float x0 = s0[r] * SCALE, x1 = s1[r] * SCALE;
      float tm = fmaxf(x0, x1);
      tm = fmaxf(tm, __shfl_xor(tm, 1, 32));
      tm = fmaxf(tm, __shfl_xor(tm, 2, 32));
      tm = fmaxf(tm, __shfl_xor(tm, 4, 32));
      tm = fmaxf(tm, __shfl_xor(tm, 8, 32));
      const float mn = fmaxf(mrow[r], tm);
      const float al = __expf(mrow[r] - mn);
      const float e0 = __expf(x0 - mn);
      const float e1 = __expf(x1 - mn);
      float rs = e0 + e1;
      rs += __shfl_xor(rs, 1, 32);
      rs += __shfl_xor(rs, 2, 32);
      rs += __shfl_xor(rs, 4, 32);
      rs += __shfl_xor(rs, 8, 32);
      lrow[r] = lrow[r] * al + rs;
      mrow[r] = mn;
      s0[r] = e0;
      s1[r] = e1;
#pragma unroll
      for (int f = 0; f < 4; ++f) oacc[f][r] *= al;
    }

    // --- transpose P (C-layout) -> A-fragment layout via per-wave LDS ---
    uint16_t* pw = &lP[wave][0][0];
#pragma unroll
    for (int r = 0; r < 8; ++r) {
      const int m = r + hi * 8;
      pw[m * 32 + n]      = f32_to_bf16(s0[r]);
      pw[m * 32 + 16 + n] = f32_to_bf16(s1[r]);
    }
    const v16bf pa = load_fragA(&lP[wave][n][0], hi);  // wave-private; DS in-order

    // --- O += P @ V (K = 32 keys, 4 d-strips of 16) ---
#pragma unroll
    for (int f = 0; f < 4; ++f) {
      const v16bf vb = load_fragB(&lVt[f * 16 + n][0], hi);
      oacc[f] = wmma_bf16(pa, vb, oacc[f]);
    }
  }

  // --- normalize and write O (bf16, [B*T, 768] with head offset) ---
#pragma unroll
  for (int r = 0; r < 8; ++r) {
    const float inv = 1.f / lrow[r];
    const int q = qbase + r + hi * 8;
    uint16_t* orow = obuf + (size_t)(b * Tq + q) * 768 + h * HD;
#pragma unroll
    for (int f = 0; f < 4; ++f)
      orow[f * 16 + n] = f32_to_bf16(oacc[f][r] * inv);
  }
}

// ---------------------------------------------------------------------------
extern "C" void kernel_launch(void* const* d_in, const int* in_sizes, int n_in,
                              void* d_out, int out_size, void* d_ws, size_t ws_size,
                              hipStream_t stream) {
  (void)in_sizes; (void)n_in; (void)out_size; (void)ws_size;
  const float* x     = (const float*)d_in[0];
  const float* Wqkv  = (const float*)d_in[1];
  const float* bqkv  = (const float*)d_in[2];
  const float* Wproj = (const float*)d_in[3];
  const float* bproj = (const float*)d_in[4];
  float* out = (float*)d_out;

  constexpr int M = 8 * 1024;   // B*T
  constexpr int D = 768, D3 = 2304;

  uint16_t* xb     = (uint16_t*)d_ws;                 // [M, 768]    12.6 MB
  uint16_t* wqkvt  = xb + (size_t)M * D;              // [2304, 768]  3.5 MB
  uint16_t* wprojt = wqkvt + (size_t)D3 * D;          // [768, 768]   1.2 MB
  uint16_t* qkvb   = wprojt + (size_t)D * D;          // [M, 2304]   37.7 MB
  uint16_t* obuf   = qkvb + (size_t)M * D3;           // [M, 768]    12.6 MB

  convert_f32_bf16<<<(M * D / 4 + 255) / 256, 256, 0, stream>>>(x, xb, M * D / 4);
  transpose_f32_bf16<<<dim3(D3 / 32, D / 32), 256, 0, stream>>>(Wqkv, wqkvt, D, D3);
  transpose_f32_bf16<<<dim3(D / 32, D / 32), 256, 0, stream>>>(Wproj, wprojt, D, D);

  gemm_bias_wmma_tdm<uint16_t>
      <<<dim3(M / 128, D3 / 64), 256, 0, stream>>>(xb, wqkvt, bqkv, qkvb, M, D3, D);
  flash_attn_wmma<<<dim3(8, 12, 8), 256, 0, stream>>>(qkvb, obuf);
  gemm_bias_wmma_tdm<float>
      <<<dim3(M / 128, D / 64), 256, 0, stream>>>(obuf, wprojt, bproj, out, M, D, D);
}
